// MultiHeadedAttention_27290222199252
// MI455X (gfx1250) — matrix operand order repaired and checked by reading, not compiled, not run on hardware
//
#include <hip/hip_runtime.h>
#include <hip/hip_bf16.h>
#include <math.h>
#include <stdint.h>

// ---------------------------------------------------------------------------
// MultiHeadedAttention forward for MI455X (gfx1250, wave32, WMMA bf16).
// All GEMMs use v_wmma_f32_16x16x32_bf16 (f32 accumulate). attn kept in f32.
// Mask is applied inside the softmax pass (coalesced byte reads) so the
// scores GEMM epilogue is a pure store stream.
// ---------------------------------------------------------------------------

typedef __attribute__((ext_vector_type(8)))  __bf16 v8bf;
typedef __attribute__((ext_vector_type(16))) __bf16 v16bf;
typedef __attribute__((ext_vector_type(8)))  float  v8f;
typedef __attribute__((ext_vector_type(4)))  float  v4f;

#define T_SEQ  2048
#define FEAT   512
#define NHEAD  8
#define DK     64
#define BSZ    4
#define MROWS  (BSZ * T_SEQ)   // 8192 tokens

static __device__ __forceinline__ v16bf frag_cat(v8bf lo, v8bf hi) {
    return __builtin_shufflevector(lo, hi, 0,1,2,3,4,5,6,7,8,9,10,11,12,13,14,15);
}

// A fragment: 16x32 (MxK) bf16.  lane: m = lane&15, khalf = lane>>4.
// elems 0-7  -> K = k0 + khalf*8 + e
// elems 8-15 -> K = k0 + 16 + khalf*8 + (e-8)
static __device__ __forceinline__ v16bf load_fragA(const __bf16* __restrict__ base,
                                                   int ld, int row0, int k0, int lane) {
    const __bf16* p = base + (size_t)(row0 + (lane & 15)) * ld + (k0 + ((lane >> 4) << 3));
    v8bf lo = *(const v8bf*)(p);
    v8bf hi = *(const v8bf*)(p + 16);
    return frag_cat(lo, hi);
}

// B fragment (NT form): B[k][n] = Wrows[n][k].  lane: n = lane&15, khalf = lane>>4.
// elems 0-7 -> K = k0 + khalf*8 + e; elems 8-15 -> K = k0 + 16 + khalf*8 + (e-8)
static __device__ __forceinline__ v16bf load_fragB(const __bf16* __restrict__ base,
                                                   int ld, int n0, int k0, int lane) {
    const __bf16* p = base + (size_t)(n0 + (lane & 15)) * ld + (k0 + ((lane >> 4) << 3));
    v8bf lo = *(const v8bf*)(p);
    v8bf hi = *(const v8bf*)(p + 16);
    return frag_cat(lo, hi);
}

// A fragment from f32 source (attention probabilities), converted to bf16 RNE.
static __device__ __forceinline__ v16bf load_fragA_f32(const float* __restrict__ base,
                                                       int ld, int row0, int k0, int lane) {
    const float* p = base + (size_t)(row0 + (lane & 15)) * ld + (k0 + ((lane >> 4) << 3));
    v16bf r;
#pragma unroll
    for (int i = 0; i < 8; ++i) r[i] = (__bf16)p[i];
#pragma unroll
    for (int i = 0; i < 8; ++i) r[i + 8] = (__bf16)p[16 + i];
    return r;
}

static __device__ __forceinline__ v8f wmma_bf16(v16bf a, v16bf b, v8f c) {
    return __builtin_amdgcn_wmma_f32_16x16x32_bf16(false, a, false, b, (short)0, c, false, false);
}

// ---------------------------------------------------------------------------
// K0: convert the four 512x512 f32 weight matrices to bf16 (contiguous dst).
// ---------------------------------------------------------------------------
__global__ __launch_bounds__(256) void cvt_weights(const float* __restrict__ a,
                                                   const float* __restrict__ b,
                                                   const float* __restrict__ c,
                                                   const float* __restrict__ d,
                                                   __bf16* __restrict__ out) {
    size_t i = (size_t)blockIdx.x * 256 + threadIdx.x;   // 4 * 262144 total
    const size_t W = (size_t)FEAT * FEAT;
    float v;
    if      (i <     W) v = a[i];
    else if (i < 2 * W) v = b[i - W];
    else if (i < 3 * W) v = c[i - 2 * W];
    else                v = d[i - 3 * W];
    out[i] = (__bf16)v;
}

// ---------------------------------------------------------------------------
// K1: LayerNorm over F=512 per token, output bf16. One 256-thread block/row.
// ---------------------------------------------------------------------------
__global__ __launch_bounds__(256) void layernorm_bf16(const float* __restrict__ x,
                                                      const float* __restrict__ gamma,
                                                      const float* __restrict__ beta,
                                                      __bf16* __restrict__ xn) {
    __shared__ float ssum[8], ssq[8];
    const float* row = x + (size_t)blockIdx.x * FEAT;
    int t = threadIdx.x;
    float a0 = row[t], a1 = row[t + 256];
    float s = a0 + a1, sq = a0 * a0 + a1 * a1;
#pragma unroll
    for (int off = 16; off; off >>= 1) {
        s  += __shfl_xor(s,  off, 32);
        sq += __shfl_xor(sq, off, 32);
    }
    int wid = t >> 5;
    if ((t & 31) == 0) { ssum[wid] = s; ssq[wid] = sq; }
    __syncthreads();
    float S = 0.f, SQ = 0.f;
#pragma unroll
    for (int i = 0; i < 8; ++i) { S += ssum[i]; SQ += ssq[i]; }
    float mu  = S * (1.0f / FEAT);
    float var = SQ * (1.0f / FEAT) - mu * mu;
    float rs  = rsqrtf(var + 1e-5f);
    __bf16* orow = xn + (size_t)blockIdx.x * FEAT;
    orow[t]       = (__bf16)((a0 - mu) * rs * gamma[t]       + beta[t]);
    orow[t + 256] = (__bf16)((a1 - mu) * rs * gamma[t + 256] + beta[t + 256]);
}

// ---------------------------------------------------------------------------
// K2: generic NT GEMM  C[m,n] = sum_k A[m,k] * W[n,k],  M=8192, N=512, K=512.
// Block = 4 waves (128 thr), block tile 64(M)x128(N), wave tile 32x64.
// MODE 0: store bf16 as q/k in [B,H,T,DK]
// MODE 1: store bf16 as vT  in [B,H,DK,T]   (transposed so attn@V is NT)
// MODE 2: store f32 row-major (final output projection)
// ---------------------------------------------------------------------------
template<int MODE>
__global__ __launch_bounds__(128) void gemm512_nt(const __bf16* __restrict__ A,
                                                  const __bf16* __restrict__ W,
                                                  void* __restrict__ outp) {
    const int K = FEAT;
    int lane = threadIdx.x & 31, wave = threadIdx.x >> 5;
    int m0 = blockIdx.y * 64 + (wave & 1) * 32;
    int n0 = blockIdx.x * 128 + (wave >> 1) * 64;

    v8f acc[2][4] = {};
    for (int k0 = 0; k0 < K; k0 += 32) {
        v16bf af[2], bf[4];
#pragma unroll
        for (int mi = 0; mi < 2; ++mi) af[mi] = load_fragA(A, K, m0 + mi * 16, k0, lane);
#pragma unroll
        for (int ni = 0; ni < 4; ++ni) bf[ni] = load_fragB(W, K, n0 + ni * 16, k0, lane);
#pragma unroll
        for (int mi = 0; mi < 2; ++mi)
#pragma unroll
            for (int ni = 0; ni < 4; ++ni)
                acc[mi][ni] = wmma_bf16(af[mi], bf[ni], acc[mi][ni]);
    }
#pragma unroll
    for (int mi = 0; mi < 2; ++mi)
#pragma unroll
        for (int ni = 0; ni < 4; ++ni) {
            int n  = n0 + ni * 16 + (lane & 15);
            int mb = m0 + mi * 16 + ((lane >> 4) << 3);
#pragma unroll
            for (int r = 0; r < 8; ++r) {
                int M = mb + r, N = n;
                float v = acc[mi][ni][r];
                if (MODE == 2) {
                    ((float*)outp)[(size_t)M * FEAT + N] = v;
                } else {
                    int b_ = M >> 11, t = M & (T_SEQ - 1);
                    int h  = N >> 6,  d = N & (DK - 1);
                    if (MODE == 0)
                        ((__bf16*)outp)[(((size_t)(b_ * NHEAD + h) * T_SEQ + t) << 6) + d] = (__bf16)v;
                    else
                        ((__bf16*)outp)[(((size_t)(b_ * NHEAD + h) * DK + d) * T_SEQ) + t] = (__bf16)v;
                }
            }
        }
}

// ---------------------------------------------------------------------------
// K3: scores = (Q K^T) / sqrt(DK), f32 into the attn output region.
// Pure GEMM; masking is deferred to the softmax pass.
// Per (b,h): M=N=2048, K=64. grid (16,32,32), wave tile 32x64.
// ---------------------------------------------------------------------------
__global__ __launch_bounds__(128) void scores_kernel(const __bf16* __restrict__ Q,
                                                     const __bf16* __restrict__ Kmat,
                                                     float* __restrict__ attn) {
    int lane = threadIdx.x & 31, wave = threadIdx.x >> 5;
    int bh = blockIdx.z;
    int m0 = blockIdx.y * 64 + (wave & 1) * 32;
    int n0 = blockIdx.x * 128 + (wave >> 1) * 64;
    const __bf16* Qb = Q    + (size_t)bh * T_SEQ * DK;
    const __bf16* Kb = Kmat + (size_t)bh * T_SEQ * DK;
    float* ob = attn + (size_t)bh * T_SEQ * T_SEQ;

    v16bf af[2][2];
#pragma unroll
    for (int mi = 0; mi < 2; ++mi)
#pragma unroll
        for (int kf = 0; kf < 2; ++kf)
            af[mi][kf] = load_fragA(Qb, DK, m0 + mi * 16, kf * 32, lane);

#pragma unroll
    for (int ni = 0; ni < 4; ++ni) {
        v16bf b0 = load_fragB(Kb, DK, n0 + ni * 16, 0,  lane);
        v16bf b1 = load_fragB(Kb, DK, n0 + ni * 16, 32, lane);
#pragma unroll
        for (int mi = 0; mi < 2; ++mi) {
            v8f acc = {};
            acc = wmma_bf16(af[mi][0], b0, acc);
            acc = wmma_bf16(af[mi][1], b1, acc);
            int n   = n0 + ni * 16 + (lane & 15);
            int mb_ = m0 + mi * 16 + ((lane >> 4) << 3);
#pragma unroll
            for (int r = 0; r < 8; ++r)
                ob[(size_t)(mb_ + r) * T_SEQ + n] = acc[r] * 0.125f;   // 1/sqrt(64)
        }
    }
}

// ---------------------------------------------------------------------------
// K4: masked softmax in place over rows of 2048 (B*H*T = 65536 rows).
// Each thread owns 8 contiguous elements: 2x float4 loads/stores + 2x dword
// mask loads per thread (fully coalesced, 1 KB per wave per access).
// ---------------------------------------------------------------------------
__global__ __launch_bounds__(256) void softmax_rows(float* __restrict__ attn,
                                                    const unsigned char* __restrict__ mask) {
    __shared__ float sred[8];
    size_t rowid = blockIdx.x;                 // [b][h][q]
    int q  = (int)(rowid & (T_SEQ - 1));
    int b_ = (int)(rowid >> 14);               // / (NHEAD*T_SEQ)
    float* row = attn + rowid * T_SEQ;
    const uint32_t* mrow =
        (const uint32_t*)(mask + ((size_t)b_ * T_SEQ + q) * T_SEQ);

    int t = threadIdx.x, wid = t >> 5;
    int base = t * 8;

    v4f x0 = *(const v4f*)(row + base);
    v4f x1 = *(const v4f*)(row + base + 4);
    uint32_t mk0 = mrow[t * 2], mk1 = mrow[t * 2 + 1];

    float v[8];
#pragma unroll
    for (int i = 0; i < 4; ++i) v[i]     = ((mk0 >> (i * 8)) & 0xFF) ? -__builtin_inff() : x0[i];
#pragma unroll
    for (int i = 0; i < 4; ++i) v[i + 4] = ((mk1 >> (i * 8)) & 0xFF) ? -__builtin_inff() : x1[i];

    float m = -__builtin_inff();
#pragma unroll
    for (int i = 0; i < 8; ++i) m = fmaxf(m, v[i]);
#pragma unroll
    for (int off = 16; off; off >>= 1) m = fmaxf(m, __shfl_xor(m, off, 32));
    if ((t & 31) == 0) sred[wid] = m;
    __syncthreads();
    float rm = sred[0];
#pragma unroll
    for (int i = 1; i < 8; ++i) rm = fmaxf(rm, sred[i]);
    __syncthreads();

    float s = 0.f;
#pragma unroll
    for (int i = 0; i < 8; ++i) { v[i] = __expf(v[i] - rm); s += v[i]; }
#pragma unroll
    for (int off = 16; off; off >>= 1) s += __shfl_xor(s, off, 32);
    if ((t & 31) == 0) sred[wid] = s;
    __syncthreads();
    float tot = 0.f;
#pragma unroll
    for (int i = 0; i < 8; ++i) tot += sred[i];
    float inv = 1.0f / tot;

    v4f o0, o1;
#pragma unroll
    for (int i = 0; i < 4; ++i) { o0[i] = v[i] * inv; o1[i] = v[i + 4] * inv; }
    *(v4f*)(row + base)     = o0;
    *(v4f*)(row + base + 4) = o1;
}

// ---------------------------------------------------------------------------
// K5: ctx = attn @ V.  Per (b,h): M=2048, N=64, K=2048.
// A = attn f32 (converted to bf16 fragments on the fly), B = vT (NT layout).
// grid (16, 32): x = M/128, y = b*h. ctx stored bf16 [token, h*64+d].
// ---------------------------------------------------------------------------
__global__ __launch_bounds__(128) void attnv_kernel(const float* __restrict__ attn,
                                                    const __bf16* __restrict__ vT,
                                                    __bf16* __restrict__ ctx) {
    int lane = threadIdx.x & 31, wave = threadIdx.x >> 5;
    int bh = blockIdx.y;
    int m0 = blockIdx.x * 128 + wave * 32;
    const float*  Ab = attn + (size_t)bh * T_SEQ * T_SEQ;
    const __bf16* Vb = vT   + (size_t)bh * DK * T_SEQ;
    int b_ = bh >> 3, h = bh & 7;

    v8f acc[2][4] = {};
    for (int k0 = 0; k0 < T_SEQ; k0 += 32) {
        v16bf af[2], bf[4];
#pragma unroll
        for (int mi = 0; mi < 2; ++mi) af[mi] = load_fragA_f32(Ab, T_SEQ, m0 + mi * 16, k0, lane);
#pragma unroll
        for (int ni = 0; ni < 4; ++ni) bf[ni] = load_fragB(Vb, T_SEQ, ni * 16, k0, lane);
#pragma unroll
        for (int mi = 0; mi < 2; ++mi)
#pragma unroll
            for (int ni = 0; ni < 4; ++ni)
                acc[mi][ni] = wmma_bf16(af[mi], bf[ni], acc[mi][ni]);
    }
#pragma unroll
    for (int mi = 0; mi < 2; ++mi)
#pragma unroll
        for (int ni = 0; ni < 4; ++ni) {
            int n  = ni * 16 + (lane & 15);
            int mb = m0 + mi * 16 + ((lane >> 4) << 3);
#pragma unroll
            for (int r = 0; r < 8; ++r)
                ctx[(size_t)(b_ * T_SEQ + mb + r) * FEAT + h * DK + n] = (__bf16)acc[mi][ni][r];
        }
}

// ---------------------------------------------------------------------------
extern "C" void kernel_launch(void* const* d_in, const int* in_sizes, int n_in,
                              void* d_out, int out_size, void* d_ws, size_t ws_size,
                              hipStream_t stream) {
    (void)in_sizes; (void)n_in; (void)out_size; (void)ws_size;
    const float*         query = (const float*)d_in[0];
    const unsigned char* mask  = (const unsigned char*)d_in[1];
    const float*         gamma = (const float*)d_in[2];
    const float*         beta  = (const float*)d_in[3];
    const float*         Wq    = (const float*)d_in[4];
    const float*         Wk    = (const float*)d_in[5];
    const float*         Wv    = (const float*)d_in[6];
    const float*         Wo    = (const float*)d_in[7];

    // workspace layout (bf16 elements), ~44 MB total
    __bf16* ws  = (__bf16*)d_ws;
    __bf16* xn  = ws;                                   // 8192*512
    __bf16* wqb = xn  + (size_t)MROWS * FEAT;           // 4x 512*512 contiguous
    __bf16* wkb = wqb + (size_t)FEAT * FEAT;
    __bf16* wvb = wkb + (size_t)FEAT * FEAT;
    __bf16* wob = wvb + (size_t)FEAT * FEAT;
    __bf16* qb  = wob + (size_t)FEAT * FEAT;            // [B,H,T,DK]
    __bf16* kb  = qb  + (size_t)BSZ * NHEAD * T_SEQ * DK;
    __bf16* vT  = kb  + (size_t)BSZ * NHEAD * T_SEQ * DK;  // [B,H,DK,T]
    __bf16* ctx = vT  + (size_t)BSZ * NHEAD * T_SEQ * DK;  // [8192,512]

    float* attn = (float*)d_out + (size_t)MROWS * FEAT;    // output #2 region

    cvt_weights<<<4096, 256, 0, stream>>>(Wq, Wk, Wv, Wo, wqb);
    layernorm_bf16<<<MROWS, 256, 0, stream>>>(query, gamma, beta, xn);

    dim3 gProj(4, 128);                                    // N/128 x M/64
    gemm512_nt<0><<<gProj, 128, 0, stream>>>(xn, wqb, (void*)qb);
    gemm512_nt<0><<<gProj, 128, 0, stream>>>(xn, wkb, (void*)kb);
    gemm512_nt<1><<<gProj, 128, 0, stream>>>(xn, wvb, (void*)vT);

    dim3 gScore(16, 32, 32);                               // N/128 x M/64 x (B*H)
    scores_kernel<<<gScore, 128, 0, stream>>>(qb, kb, attn);

    softmax_rows<<<BSZ * NHEAD * T_SEQ, 256, 0, stream>>>(attn, mask);

    dim3 gAV(16, 32);                                      // M/128 x (B*H)
    attnv_kernel<<<gAV, 128, 0, stream>>>(attn, vT, ctx);

    gemm512_nt<2><<<gProj, 128, 0, stream>>>(ctx, wob, d_out);
}
